// AttentionGCN_20005957665495
// MI455X (gfx1250) — matrix operand order repaired and checked by reading, not compiled, not run on hardware
//
#include <hip/hip_runtime.h>
#include <hip/hip_bf16.h>

#define HH   4
#define FF   32
#define DIN  128
#define DOUT 128   // HH*FF

typedef __attribute__((ext_vector_type(16))) _Float16 v16h;
typedef __attribute__((ext_vector_type(8)))  float    v8f;

#define LDS_PITCH 132                       // 16B-aligned, bank-conflict-free
#define WPACK_ELEMS (8 * 4 * 32 * 16)       // col-tiles * k-steps * lanes * halves

// ---- monotone float <-> unsigned encoding for atomicMax-based segment max ----
__device__ __forceinline__ unsigned enc_f32(float f) {
    unsigned u = __float_as_uint(f);
    return (u & 0x80000000u) ? ~u : (u | 0x80000000u);
}
__device__ __forceinline__ float dec_f32(unsigned u) {
    unsigned b = (u & 0x80000000u) ? (u & 0x7FFFFFFFu) : ~u;
    return __uint_as_float(b);
}

// -------- zero-init scratch (segmax enc=0 is below enc of any real float) ----
__global__ void gat_init_zero(unsigned* __restrict__ p, int n) {
    int i = blockIdx.x * blockDim.x + threadIdx.x;
    if (i < n) p[i] = 0u;
}

// -------- K0: pre-pack W into f16 WMMA B-fragment layout ---------------------
// wp[((ct*4 + ks)*32 + lane)*16 + j] = f16(W[(ks*32 + 8*(lane>>4) + (j&7) + 2*(j&8))*128 + ct*16 + (lane&15)])
// B layout: lanes 0-15 hold K=k0+{0..7,16..23}, lanes 16-31 hold K=k0+{8..15,24..31} (2 halves/VGPR).
__global__ void gat_pack_w(const float* __restrict__ W, _Float16* __restrict__ wp) {
    int i = blockIdx.x * blockDim.x + threadIdx.x;
    if (i >= WPACK_ELEMS) return;
    int j    = i & 15;
    int lane = (i >> 4) & 31;
    int ks   = (i >> 9) & 3;
    int ct   = i >> 11;
    int k    = ks * 32 + ((lane & 16) ? 8 : 0) + (j & 7) + ((j & 8) ? 16 : 0);
    int col  = ct * 16 + (lane & 15);
    wp[i] = (_Float16)W[(size_t)k * DOUT + col];
}

// -------- K1: h = x @ W via v_wmma_f32_16x16x32_f16 -------------------------
// grid.x = N/16 row tiles; 256 threads = 8 waves; wave w owns column tile w.
// x tile staged into LDS once per block via global_load_async_to_lds_b128.
__global__ void gat_proj_wmma(const float* __restrict__ x,
                              const _Float16* __restrict__ wp,
                              float* __restrict__ h, int n) {
    __shared__ float ldsx[16 * LDS_PITCH];   // 16 rows x 128 f32, pitch 132

    const int wave = threadIdx.x >> 5;       // 0..7 -> column tile
    const int lane = threadIdx.x & 31;
    const int row0 = blockIdx.x * 16;
    const int col0 = wave * 16;
    const int m    = lane & 15;

    // ---- async-stage x[row0:row0+16, 0:128] into LDS (8 KB, 512 x 16B) -----
    {
        const int t = threadIdx.x;
#pragma unroll
        for (int i = 0; i < 2; ++i) {
            int c = t + i * 256;             // chunk id 0..511
            int r = c >> 5;                  // row in tile
            int q = c & 31;                  // 16B chunk within row
            int gr = row0 + r; if (gr >= n) gr = n - 1;
            const float* gp = x + (size_t)gr * DIN + q * 4;
            unsigned la = (unsigned)(size_t)ldsx + (unsigned)(r * LDS_PITCH * 4 + q * 16);
            asm volatile("global_load_async_to_lds_b128 %0, %1, off"
                         :: "v"(la), "v"(gp) : "memory");
        }
        asm volatile("s_wait_asynccnt 0x0" ::: "memory");
        __syncthreads();
    }

    const float* lr = ldsx + m * LDS_PITCH;  // A row for this lane (both halves)

    v8f acc = {};
#pragma unroll
    for (int k0 = 0; k0 < DIN; k0 += 32) {
        // A tile 16x32 f16: lanes 0-15 K={0..7,16..23}, lanes 16-31 K={8..15,24..31}
        const int ka = k0 + ((lane & 16) ? 8 : 0);
        v16h a;
#pragma unroll
        for (int j = 0; j < 8; ++j) {
            a[j]     = (_Float16)lr[ka + j];
            a[j + 8] = (_Float16)lr[ka + 16 + j];
        }
        // B fragment: pre-packed contiguous 16 halves per lane
        const v16h b = *(const v16h*)(wp + ((size_t)(wave * 4 + (k0 >> 5)) * 32 + lane) * 16);

        acc = __builtin_amdgcn_wmma_f32_16x16x32_f16(
            /*neg_a=*/false, a, /*neg_b=*/false, b,
            /*c_mod=*/(short)0, acc, /*reuse_a=*/false, /*reuse_b=*/false);
    }
    // D layout: lane -> col (lane&15); VGPR r -> row r (+8 for lanes 16-31)
    const int colD  = col0 + m;
    const int rbase = (lane & 16) ? 8 : 0;
#pragma unroll
    for (int r = 0; r < 8; ++r) {
        int rw = row0 + rbase + r;
        if (rw < n) h[(size_t)rw * DOUT + colD] = acc[r];
    }
}

// -------- K2: a_src[n,h], a_dst[n,h] ----------------------------------------
__global__ void gat_att_scores(const float* __restrict__ h,
                               const float* __restrict__ att_src,
                               const float* __restrict__ att_dst,
                               float* __restrict__ a_src,
                               float* __restrict__ a_dst, int n) {
    int idx = blockIdx.x * blockDim.x + threadIdx.x;   // node*HH + head
    if (idx >= n * HH) return;
    int node = idx >> 2, hd = idx & 3;
    const float* hp = h + (size_t)node * DOUT + hd * FF;
    const float* as = att_src + hd * FF;
    const float* ad = att_dst + hd * FF;
    float s = 0.f, d = 0.f;
#pragma unroll
    for (int f = 0; f < FF; ++f) { float v = hp[f]; s += v * as[f]; d += v * ad[f]; }
    a_src[idx] = s;
    a_dst[idx] = d;
}

__device__ __forceinline__ void edge_sd(const int* __restrict__ ei, int e,
                                        int e_raw, int& s, int& d) {
    if (e < e_raw) { s = ei[e]; d = ei[e_raw + e]; }
    else           { s = d = e - e_raw; }              // self loop
}

// -------- K3a: per-(edge,head) leaky-relu logit -> segment max --------------
__global__ void gat_edge_max(const int* __restrict__ ei,
                             const float* __restrict__ a_src,
                             const float* __restrict__ a_dst,
                             unsigned* __restrict__ segmax,
                             int e_raw, int n) {
    int idx = blockIdx.x * blockDim.x + threadIdx.x;   // (edge)*HH + head
    int total = (e_raw + n) * HH;
    if (idx >= total) return;
    int e = idx >> 2, hd = idx & 3;
    int s, d; edge_sd(ei, e, e_raw, s, d);
    float l = a_src[s * HH + hd] + a_dst[d * HH + hd];
    l = (l > 0.f) ? l : 0.2f * l;
    atomicMax(&segmax[d * HH + hd], enc_f32(l));
}

// -------- K3b: e = exp(logit - max); denom += e; acc[dst] += e * h[src] -----
__global__ void gat_edge_accum(const int* __restrict__ ei,
                               const float* __restrict__ a_src,
                               const float* __restrict__ a_dst,
                               const unsigned* __restrict__ segmax,
                               const float* __restrict__ h,
                               float* __restrict__ denom,
                               float* __restrict__ acc,
                               int e_raw, int n) {
    int idx = blockIdx.x * blockDim.x + threadIdx.x;
    int total = (e_raw + n) * HH;
    if (idx >= total) return;
    int e = idx >> 2, hd = idx & 3;
    int s, d; edge_sd(ei, e, e_raw, s, d);
    float l = a_src[s * HH + hd] + a_dst[d * HH + hd];
    l = (l > 0.f) ? l : 0.2f * l;
    float ex = __expf(l - dec_f32(segmax[d * HH + hd]));
    atomicAdd(&denom[d * HH + hd], ex);

    const float* hp = h + (size_t)s * DOUT + hd * FF;
    float*       op = acc + (size_t)d * DOUT + hd * FF;
    __builtin_prefetch(hp, 0, 1);                      // global_prefetch_b8
#pragma unroll
    for (int f = 0; f < FF; ++f)
        atomicAdd(&op[f], ex * hp[f]);
}

// -------- K4: normalize, +bias, relu, linear head, per-node reduce ----------
__global__ void gat_finalize(const float* __restrict__ acc,
                             const float* __restrict__ denom,
                             const float* __restrict__ gat_bias,
                             const float* __restrict__ lin_W,
                             const float* __restrict__ lin_b,
                             float* __restrict__ out, int n) {
    __shared__ float red[HH];
    const int node = blockIdx.x;
    const int t    = threadIdx.x;                      // 0..127 -> (head, f)
    const float den = denom[node * HH + (t >> 5)];
    float v = acc[(size_t)node * DOUT + t] / den + gat_bias[t];
    v = fmaxf(v, 0.f);
    float p = v * lin_W[t];
#pragma unroll
    for (int off = 16; off > 0; off >>= 1)             // wave32 reduction
        p += __shfl_down(p, off, 32);
    if ((t & 31) == 0) red[t >> 5] = p;
    __syncthreads();
    if (t == 0) out[node] = red[0] + red[1] + red[2] + red[3] + lin_b[0];
}

extern "C" void kernel_launch(void* const* d_in, const int* in_sizes, int n_in,
                              void* d_out, int out_size, void* d_ws, size_t ws_size,
                              hipStream_t stream) {
    const float* x        = (const float*)d_in[0];
    const int*   ei       = (const int*)  d_in[1];
    const float* W        = (const float*)d_in[2];
    const float* att_src  = (const float*)d_in[3];
    const float* att_dst  = (const float*)d_in[4];
    const float* gat_bias = (const float*)d_in[5];
    const float* lin_W    = (const float*)d_in[6];
    const float* lin_b    = (const float*)d_in[7];

    const int N = in_sizes[0] / DIN;
    const int E = in_sizes[1] / 2;

    // workspace layout
    float*    h      = (float*)d_ws;                         // N*128
    float*    a_src  = h     + (size_t)N * DOUT;             // N*4
    float*    a_dst  = a_src + (size_t)N * HH;               // N*4
    unsigned* segmax = (unsigned*)(a_dst + (size_t)N * HH);  // N*4
    float*    denom  = (float*)(segmax + (size_t)N * HH);    // N*4
    float*    acc    = denom + (size_t)N * HH;               // N*128
    _Float16* wpack  = (_Float16*)(acc + (size_t)N * DOUT);  // 16384 halves (32 KB)

    // zero segmax|denom|acc (contiguous): N*(4+4+128) dwords
    const int zelems = N * (HH + HH + DOUT);
    gat_init_zero<<<(zelems + 255) / 256, 256, 0, stream>>>(segmax, zelems);

    gat_pack_w<<<(WPACK_ELEMS + 255) / 256, 256, 0, stream>>>(W, wpack);
    gat_proj_wmma<<<(N + 15) / 16, 256, 0, stream>>>(x, wpack, h, N);
    gat_att_scores<<<(N * HH + 255) / 256, 256, 0, stream>>>(h, att_src, att_dst,
                                                             a_src, a_dst, N);
    const int tot = (E + N) * HH;
    gat_edge_max<<<(tot + 255) / 256, 256, 0, stream>>>(ei, a_src, a_dst, segmax, E, N);
    gat_edge_accum<<<(tot + 255) / 256, 256, 0, stream>>>(ei, a_src, a_dst, segmax,
                                                          h, denom, acc, E, N);
    gat_finalize<<<N, DOUT, 0, stream>>>(acc, denom, gat_bias, lin_W, lin_b,
                                         (float*)d_out, N);
}
